// Block_67113158967580
// MI455X (gfx1250) — matrix operand order repaired and checked by reading, not compiled, not run on hardware
//
#include <hip/hip_runtime.h>
#include <hip/hip_bf16.h>

typedef unsigned short u16;
typedef __attribute__((ext_vector_type(16))) __bf16       v16bf;
typedef __attribute__((ext_vector_type(8)))  float        v8f;
typedef __attribute__((ext_vector_type(8)))  unsigned int v8u;
typedef __attribute__((ext_vector_type(4)))  unsigned int u32x4;
typedef __attribute__((ext_vector_type(8)))  int          i32x8;
typedef __attribute__((ext_vector_type(4)))  int          i32x4;

#define B_   2
#define T_   2048
#define C_   2048
#define H_   16
#define G_   4
#define D_   128
#define QKV_ 3072
#define FFN_ 8192

#if __has_builtin(__builtin_amdgcn_tensor_load_to_lds)
#define HAS_TDM 1
#else
#define HAS_TDM 0
#endif

__device__ __forceinline__ u16 f2bf(float f) {
  unsigned int u = __builtin_bit_cast(unsigned int, f);
  u += 0x7FFFu + ((u >> 16) & 1u);     // round to nearest even
  return (u16)(u >> 16);
}

// Build a 16-element bf16 fragment from two 8-half (16B) chunks.
__device__ __forceinline__ v16bf load_frag(const u16* p0, const u16* p1) {
  uint4 a = *(const uint4*)p0;
  uint4 b = *(const uint4*)p1;
  v8u u;
  u[0] = a.x; u[1] = a.y; u[2] = a.z; u[3] = a.w;
  u[4] = b.x; u[5] = b.y; u[6] = b.z; u[7] = b.w;
  return __builtin_bit_cast(v16bf, u);
}

// --- CDNA5 async global->LDS (ASYNCcnt) ------------------------------------
// GVS form: mem = SADDR(base) + VADDR(byte offset); VDST = per-lane LDS addr.
__device__ __forceinline__ void async_ld_b128(unsigned ldsoff, const u16* base,
                                              unsigned byteoff) {
  asm volatile("global_load_async_to_lds_b128 %0, %1, %2"
               :: "v"(ldsoff), "v"(byteoff), "s"(base) : "memory");
}
template <int N> __device__ __forceinline__ void wait_async() {
  asm volatile("s_wait_asynccnt %0" :: "n"(N) : "memory");
}

// --- CDNA5 Tensor Data Mover: 2-D tile (ROWSxCOLS bf16, row stride K elems)
#if HAS_TDM
__device__ __forceinline__ void tdm_load_tile(const u16* gsrc, unsigned ldsoff,
                                              int K, int rows, int cols) {
  unsigned long long ga = (unsigned long long)gsrc;
  u32x4 g0;
  g0[0] = 1u;                                  // count=1, user mode
  g0[1] = ldsoff;                              // lds_addr (bytes)
  g0[2] = (unsigned)ga;                        // global_addr[31:0]
  g0[3] = (unsigned)(ga >> 32) | (2u << 30);   // global_addr[56:32] | type=2
  i32x8 g1;
  g1[0] = 0x00010000;                          // data_size = 2 bytes
  g1[1] = cols << 16;                          // tensor_dim0[15:0]
  g1[2] = rows << 16;                          // tensor_dim1[15:0]
  g1[3] = cols << 16;                          // tile_dim0
  g1[4] = rows;                                // tile_dim1 (tile_dim2 = 0)
  g1[5] = K;                                   // tensor_dim0_stride[31:0]
  g1[6] = 0;
  g1[7] = 0;
  i32x4 z4; z4[0] = 0; z4[1] = 0; z4[2] = 0; z4[3] = 0;
  i32x8 z8;
  for (int i = 0; i < 8; ++i) z8[i] = 0;
  // 6-arg form on this toolchain: (g0, g1, g2, g3, <zero-filled i32x8>, cpol)
  __builtin_amdgcn_tensor_load_to_lds(g0, g1, z4, z4, z8, 0);
}
#endif

// ---------------------------------------------------------------- LayerNorm
__global__ __launch_bounds__(256) void ln_dual_kernel(
    const float* __restrict__ x,
    const float* __restrict__ w1, const float* __restrict__ b1,
    const float* __restrict__ w2, const float* __restrict__ b2,
    u16* __restrict__ xn1, u16* __restrict__ xn2)
{
  __shared__ float sS[256];
  __shared__ float sQ[256];
  const int tid = threadIdx.x;
  const size_t row = blockIdx.x;
  const float* xr = x + row * C_;

  float4 a = *(const float4*)(xr + tid * 8);
  float4 c = *(const float4*)(xr + tid * 8 + 4);
  float vals[8] = {a.x, a.y, a.z, a.w, c.x, c.y, c.z, c.w};
  float s = 0.f, sq = 0.f;
  for (int i = 0; i < 8; ++i) { s += vals[i]; sq += vals[i] * vals[i]; }
  sS[tid] = s; sQ[tid] = sq;
  __syncthreads();
  for (int off = 128; off > 0; off >>= 1) {
    if (tid < off) { sS[tid] += sS[tid + off]; sQ[tid] += sQ[tid + off]; }
    __syncthreads();
  }
  const float mu  = sS[0] * (1.0f / C_);
  const float var = sQ[0] * (1.0f / C_) - mu * mu;
  const float rs  = rsqrtf(var + 1e-5f);
  for (int i = 0; i < 8; ++i) {
    const int cc = tid * 8 + i;
    const float n = (vals[i] - mu) * rs;
    xn1[row * C_ + cc] = f2bf(n * w1[cc] + b1[cc]);
    xn2[row * C_ + cc] = f2bf(n * w2[cc] + b2[cc]);
  }
}

// ------------------------------------------- fp32 [K][N] -> bf16 [N][K]
__global__ __launch_bounds__(256) void transpose_bf16_kernel(
    const float* __restrict__ in, u16* __restrict__ out, int K, int N)
{
  __shared__ float tile[32][33];
  const int kb = blockIdx.y * 32;
  const int nb = blockIdx.x * 32;
  for (int r = threadIdx.y; r < 32; r += 8)
    tile[r][threadIdx.x] = in[(size_t)(kb + r) * N + nb + threadIdx.x];
  __syncthreads();
  for (int r = threadIdx.y; r < 32; r += 8)
    out[(size_t)(nb + r) * K + kb + threadIdx.x] = f2bf(tile[threadIdx.x][r]);
}

// --------------------------------------------------------------- GEMM
// out[M,N] = A[M,K](bf16) @ Wt[N,K](bf16)^T + bias, epilogue by mode:
//   0: bf16 out     1: bf16 out + exact GELU
//   2: f32 out      3: f32 out + res1 + res2
// Double-buffered LDS; A tile via global_load_async_to_lds_b128,
// B tile via per-wave TDM tensor_load_to_lds (async fallback).
// Final iteration peeled so the steady-state loop body is branchless.
__global__ __launch_bounds__(256) void gemm_bf16_kernel(
    const u16* __restrict__ A, const u16* __restrict__ Wt,
    const float* __restrict__ bias, void* __restrict__ outp,
    const float* __restrict__ res1, const float* __restrict__ res2,
    int M, int N, int K, int mode)
{
  __shared__ u16 As[2][128 * 32];
  __shared__ u16 Bs[2][128 * 32];
  const int tid  = threadIdx.x;
  const int lane = tid & 31;
  const int wid  = tid >> 5;
  const int wm   = wid >> 1;   // 0..3  (rows)
  const int wn   = wid & 1;    // 0..1  (cols)
  const int rowbase = blockIdx.y * 128;
  const int colbase = blockIdx.x * 128;
  const int l15  = lane & 15;
  const int half = lane >> 4;
  const int mld  = tid >> 1;          // 0..127
  const int kld  = (tid & 1) * 16;    // 0 / 16
  const int uwid = __builtin_amdgcn_readfirstlane(wid);  // uniform wave id

#if HAS_TDM
  constexpr int ASY_N = 2;   // only A-tile uses ASYNCcnt (2 instr / buffer)
#else
  constexpr int ASY_N = 4;   // A + B tiles via async (4 instr / buffer)
#endif

  v8f acc[2][4];
  for (int mt = 0; mt < 2; ++mt)
    for (int nt = 0; nt < 4; ++nt)
      for (int j = 0; j < 8; ++j) acc[mt][nt][j] = 0.0f;

  auto stage = [&](int it) {
    const int kk  = it * 32;
    const int buf = it & 1;
    // A tile: 128x32 bf16, each thread two b128 async copies
    unsigned lo = (unsigned)(unsigned long long)&As[buf][mld * 32 + kld];
    unsigned go = (unsigned)(((size_t)(rowbase + mld) * K + kk + kld) * 2);
    async_ld_b128(lo, A, go);
    async_ld_b128(lo + 16, A, go + 16);
#if HAS_TDM
    // B tile: each wave DMAs a 16x32 slice via the Tensor Data Mover
    tdm_load_tile(Wt + (size_t)(colbase + uwid * 16) * K + kk,
                  (unsigned)(unsigned long long)&Bs[buf][uwid * 16 * 32],
                  K, 16, 32);
#else
    unsigned lb = (unsigned)(unsigned long long)&Bs[buf][mld * 32 + kld];
    unsigned gb = (unsigned)(((size_t)(colbase + mld) * K + kk + kld) * 2);
    async_ld_b128(lb, Wt, gb);
    async_ld_b128(lb + 16, Wt, gb + 16);
#endif
  };

  auto compute = [&](int buf) {
    v16bf aF[2], bF[4];
    for (int mt = 0; mt < 2; ++mt) {
      const u16* p = &As[buf][(wm * 32 + mt * 16 + l15) * 32 + half * 8];
      aF[mt] = load_frag(p, p + 16);
    }
    for (int nt = 0; nt < 4; ++nt) {
      const u16* p = &Bs[buf][(wn * 64 + nt * 16 + l15) * 32 + half * 8];
      bF[nt] = load_frag(p, p + 16);
    }
    for (int mt = 0; mt < 2; ++mt)
      for (int nt = 0; nt < 4; ++nt)
        acc[mt][nt] = __builtin_amdgcn_wmma_f32_16x16x32_bf16(
            false, aF[mt], false, bF[nt], (short)0, acc[mt][nt], false, false);
  };

  const int nIters = K >> 5;   // >= 64 for all shapes here
  stage(0);
  // steady state: branchless body, next tile DMA overlaps current WMMAs
  for (int it = 0; it < nIters - 1; ++it) {
    stage(it + 1);
#if HAS_TDM
    __builtin_amdgcn_s_wait_tensorcnt(1);
#endif
    wait_async<ASY_N>();
    __syncthreads();
    compute(it & 1);
    __syncthreads();   // frag reads done before next overwrite of this buf
  }
  // drain: last tile
#if HAS_TDM
  __builtin_amdgcn_s_wait_tensorcnt(0);
#endif
  wait_async<0>();
  __syncthreads();
  compute((nIters - 1) & 1);

  for (int mt = 0; mt < 2; ++mt)
    for (int nt = 0; nt < 4; ++nt)
      for (int j = 0; j < 8; ++j) {
        const int row = rowbase + wm * 32 + mt * 16 + half * 8 + j;
        const int col = colbase + wn * 64 + nt * 16 + l15;
        const size_t idx = (size_t)row * N + col;
        float v = acc[mt][nt][j] + bias[col];
        if (mode == 1) v = 0.5f * v * (1.0f + erff(v * 0.7071067811865475f));
        if (mode == 3) v += res1[idx] + res2[idx];
        if (mode <= 1) ((u16*)outp)[idx] = f2bf(v);
        else           ((float*)outp)[idx] = v;
      }
}

// ------------------------------------------------ QKV split + RoPE
__global__ __launch_bounds__(256) void rope_split_kernel(
    const float* __restrict__ qkv, const float* __restrict__ cosb,
    const float* __restrict__ sinb,
    u16* __restrict__ q, u16* __restrict__ k, u16* __restrict__ vt)
{
  const int b = blockIdx.x / T_;
  const int t = blockIdx.x % T_;
  const float* row = qkv + (size_t)(b * T_ + t) * QKV_;
  for (int idx = threadIdx.x; idx < QKV_; idx += 256) {
    const int gg = idx / 768;
    const int j  = (idx % 768) / 128;
    const int d  = idx & 127;
    float v = row[idx];
    if (j < 5 && d < 32) {   // RoPE on q and k, first 32 dims
      const float pv = row[gg * 768 + j * 128 + (d < 16 ? d + 16 : d - 16)];
      const float cv = cosb[t * 32 + d];
      const float sv = sinb[t * 32 + d];
      v = (d < 16) ? (v * cv - pv * sv) : (v * cv + pv * sv);
    }
    if (j < 4) {             // query heads, pre-scaled by 1/sqrt(D)
      const int h = gg * 4 + j;
      q[((size_t)(b * H_ + h) * T_ + t) * D_ + d] = f2bf(v * 0.08838834764831845f);
    } else if (j == 4) {     // key
      k[((size_t)(b * G_ + gg) * T_ + t) * D_ + d] = f2bf(v);
    } else {                 // value, transposed to [D][T]
      vt[((size_t)(b * G_ + gg) * D_ + d) * T_ + t] = f2bf(v);
    }
  }
}

// ------------------------------------------------ Flash attention (1 wave)
__global__ __launch_bounds__(32) void attn_kernel(
    const u16* __restrict__ q, const u16* __restrict__ k,
    const u16* __restrict__ vt, u16* __restrict__ y)
{
  __shared__ u16 Pbuf[16 * 32];
  const int lane = threadIdx.x;
  const int l15  = lane & 15;
  const int half = lane >> 4;
  const int qb = blockIdx.x * 16;
  const int h  = blockIdx.y;
  const int b  = blockIdx.z;
  const int g  = h >> 2;
  const size_t qoff = (size_t)(b * H_ + h) * T_ * D_;
  const size_t koff = (size_t)(b * G_ + g) * T_ * D_;
  const size_t voff = (size_t)(b * G_ + g) * D_ * T_;

  v16bf qF[4];
  for (int kc = 0; kc < 4; ++kc) {
    const u16* p = q + qoff + (size_t)(qb + l15) * D_ + kc * 32 + half * 8;
    qF[kc] = load_frag(p, p + 16);
  }

  v8f O[8];
  float mrow[8], lrow[8];
  for (int dd = 0; dd < 8; ++dd)
    for (int j = 0; j < 8; ++j) O[dd][j] = 0.0f;
  for (int j = 0; j < 8; ++j) { mrow[j] = -3.0e38f; lrow[j] = 0.0f; }

  for (int kb = 0; kb < qb + 16; kb += 32) {
    v8f S[2];
    for (int sub = 0; sub < 2; ++sub)
      for (int j = 0; j < 8; ++j) S[sub][j] = 0.0f;

    for (int sub = 0; sub < 2; ++sub) {
      if (kb + sub * 16 < qb + 16) {   // uniform branch: skip fully-masked tile
        for (int kc = 0; kc < 4; ++kc) {
          const u16* p = k + koff + (size_t)(kb + sub * 16 + l15) * D_ + kc * 32 + half * 8;
          v16bf kF = load_frag(p, p + 16);
          S[sub] = __builtin_amdgcn_wmma_f32_16x16x32_bf16(
              false, qF[kc], false, kF, (short)0, S[sub], false, false);
        }
      }
    }

    float scl[8];
    for (int j = 0; j < 8; ++j) {
      const int rowg = qb + half * 8 + j;
      for (int sub = 0; sub < 2; ++sub) {
        const int colg = kb + sub * 16 + l15;
        if (colg > rowg) S[sub][j] = -3.0e38f;   // causal mask
      }
      float r = fmaxf(S[0][j], S[1][j]);
      for (int off = 1; off < 16; off <<= 1) r = fmaxf(r, __shfl_xor(r, off, 32));
      const float mnew = fmaxf(mrow[j], r);
      const float sc = __expf(mrow[j] - mnew);
      const float p0 = __expf(S[0][j] - mnew);
      const float p1 = __expf(S[1][j] - mnew);
      float sum = p0 + p1;
      for (int off = 1; off < 16; off <<= 1) sum += __shfl_xor(sum, off, 32);
      lrow[j] = lrow[j] * sc + sum;
      mrow[j] = mnew;
      scl[j]  = sc;
      Pbuf[(half * 8 + j) * 32 + l15]      = f2bf(p0);
      Pbuf[(half * 8 + j) * 32 + 16 + l15] = f2bf(p1);
    }
    for (int dd = 0; dd < 8; ++dd)
      for (int j = 0; j < 8; ++j) O[dd][j] *= scl[j];

    __syncthreads();
    const u16* pp = Pbuf + l15 * 32 + half * 8;   // C-layout -> A-fragment
    v16bf pF = load_frag(pp, pp + 16);
    __syncthreads();

    for (int dd = 0; dd < 8; ++dd) {
      const u16* p = vt + voff + (size_t)(dd * 16 + l15) * T_ + kb + half * 8;
      v16bf vF = load_frag(p, p + 16);
      O[dd] = __builtin_amdgcn_wmma_f32_16x16x32_bf16(
          false, pF, false, vF, (short)0, O[dd], false, false);
    }
  }

  for (int dd = 0; dd < 8; ++dd)
    for (int j = 0; j < 8; ++j) {
      const float v = O[dd][j] / lrow[j];
      y[(size_t)(b * T_ + qb + half * 8 + j) * C_ + h * D_ + dd * 16 + l15] = f2bf(v);
    }
}

// ----------------------------------------------------------------- launch
extern "C" void kernel_launch(void* const* d_in, const int* in_sizes, int n_in,
                              void* d_out, int out_size, void* d_ws, size_t ws_size,
                              hipStream_t stream)
{
  const float* x     = (const float*)d_in[0];
  const float* cosb  = (const float*)d_in[1];
  const float* sinb  = (const float*)d_in[2];
  const float* ln1w  = (const float*)d_in[3];
  const float* ln1b  = (const float*)d_in[4];
  const float* wqkv  = (const float*)d_in[5];
  const float* bqkv  = (const float*)d_in[6];
  const float* wproj = (const float*)d_in[7];
  const float* bproj = (const float*)d_in[8];
  const float* ln2w  = (const float*)d_in[9];
  const float* ln2b  = (const float*)d_in[10];
  const float* wfc1  = (const float*)d_in[11];
  const float* bfc1  = (const float*)d_in[12];
  const float* wfc2  = (const float*)d_in[13];
  const float* bfc2  = (const float*)d_in[14];

  char* w = (char*)d_ws;
  size_t off = 0;
  auto take = [&](size_t bytes) -> char* {
    char* p = w + off;
    off += (bytes + 255) & ~(size_t)255;
    return p;
  };
  const size_t M = (size_t)B_ * T_;                      // 4096 rows
  u16*   xn1   = (u16*)  take(M * C_ * 2);
  u16*   xn2   = (u16*)  take(M * C_ * 2);
  u16*   wqkvT = (u16*)  take((size_t)QKV_ * C_ * 2);    // [3072][2048]
  u16*   wprojT= (u16*)  take((size_t)C_ * C_ * 2);      // [2048][2048]
  u16*   wfc1T = (u16*)  take((size_t)FFN_ * C_ * 2);    // [8192][2048]
  u16*   wfc2T = (u16*)  take((size_t)C_ * FFN_ * 2);    // [2048][8192]
  float* qkvf  = (float*)take(M * QKV_ * 4);
  u16*   qb    = (u16*)  take((size_t)B_ * H_ * T_ * D_ * 2);
  u16*   kb    = (u16*)  take((size_t)B_ * G_ * T_ * D_ * 2);
  u16*   vtb   = (u16*)  take((size_t)B_ * G_ * D_ * T_ * 2);
  u16*   yb    = (u16*)  take(M * C_ * 2);
  float* hb    = (float*)take(M * C_ * 4);
  u16*   hid   = (u16*)  take(M * FFN_ * 2);
  (void)in_sizes; (void)n_in; (void)out_size; (void)ws_size;

  // 1) dual LayerNorm -> bf16
  ln_dual_kernel<<<dim3((unsigned)M), 256, 0, stream>>>(x, ln1w, ln1b, ln2w, ln2b, xn1, xn2);

  // 2) weight transpose+convert fp32 [K][N] -> bf16 [N][K]
  transpose_bf16_kernel<<<dim3(QKV_/32, C_/32), dim3(32, 8), 0, stream>>>(wqkv,  wqkvT,  C_,  QKV_);
  transpose_bf16_kernel<<<dim3(C_/32,   C_/32), dim3(32, 8), 0, stream>>>(wproj, wprojT, C_,  C_);
  transpose_bf16_kernel<<<dim3(FFN_/32, C_/32), dim3(32, 8), 0, stream>>>(wfc1,  wfc1T,  C_,  FFN_);
  transpose_bf16_kernel<<<dim3(C_/32, FFN_/32), dim3(32, 8), 0, stream>>>(wfc2,  wfc2T,  FFN_, C_);

  // 3) QKV projection (fp32 out)
  gemm_bf16_kernel<<<dim3(QKV_/128, M/128), 256, 0, stream>>>(
      xn1, wqkvT, bqkv, qkvf, nullptr, nullptr, (int)M, QKV_, C_, 2);

  // 4) RoPE + split into q / k / v^T (bf16)
  rope_split_kernel<<<dim3((unsigned)M), 256, 0, stream>>>(qkvf, cosb, sinb, qb, kb, vtb);

  // 5) causal flash attention -> y (bf16)
  attn_kernel<<<dim3(T_/16, H_, B_), 32, 0, stream>>>(qb, kb, vtb, yb);

  // 6) attention output projection (fp32 out, residual added at the end)
  gemm_bf16_kernel<<<dim3(C_/128, M/128), 256, 0, stream>>>(
      yb, wprojT, bproj, hb, nullptr, nullptr, (int)M, C_, C_, 2);

  // 7) MLP fc1 + exact GELU (bf16 out)
  gemm_bf16_kernel<<<dim3(FFN_/128, M/128), 256, 0, stream>>>(
      xn2, wfc1T, bfc1, hid, nullptr, nullptr, (int)M, FFN_, C_, 1);

  // 8) MLP fc2 + bias + attn-residual + input-residual -> d_out (fp32)
  gemm_bf16_kernel<<<dim3(C_/128, M/128), 256, 0, stream>>>(
      hid, wfc2T, bfc2, d_out, hb, x, (int)M, C_, FFN_, 3);
}
